// _CausalSelfAttention_58248346468729
// MI455X (gfx1250) — compile-verified, matrix operand order checked by reading, not run on hardware
//
#include <hip/hip_runtime.h>
#include <hip/hip_bf16.h>

// ---------------------------------------------------------------------------
// CDNA5 (gfx1250) causal self-attention: QKV GEMM + RoPE + flash attention +
// output projection. Matmuls via v_wmma_f32_16x16x32_bf16 (wave32).
// Double-buffered LDS staging; bf16 tile streams moved by the Tensor Data
// Mover (tensor_load_to_lds + s_wait_tensorcnt) when the builtin exists.
// ---------------------------------------------------------------------------

typedef __attribute__((ext_vector_type(16))) __bf16 v16bf;
typedef __attribute__((ext_vector_type(8)))  float  v8f;

#define NHEAD 16
#define HSZ   128

#if __has_builtin(__builtin_amdgcn_tensor_load_to_lds) && __has_builtin(__builtin_amdgcn_s_wait_tensorcnt)
#define HAVE_TDM 1
#else
#define HAVE_TDM 0
#endif

#if HAVE_TDM
typedef __attribute__((ext_vector_type(4))) unsigned int u32x4;
typedef __attribute__((ext_vector_type(8))) int          i32x8;
typedef __attribute__((ext_vector_type(4))) int          i32x4;

// Issue a TDM load of a 2-D bf16 tile (height x width elems, row stride in
// elems) from global memory into LDS at lds_off. Tensor dims == tile dims so
// no OOB handling is engaged. D# layout per CDNA5 ISA ch.8.
// Toolchain here uses the 6-arg builtin: (g0, g1, g2, g3, g4, cpol).
__device__ __forceinline__ void tdm_load_2d(const void* gaddr, unsigned lds_off,
                                            int width, int height, int row_stride) {
    unsigned long long ga = (unsigned long long)(uintptr_t)gaddr;
    u32x4 g0;
    g0[0] = 1u;                                              // count=1 (valid), user mode
    g0[1] = lds_off;                                         // LDS byte address
    g0[2] = (unsigned)(ga & 0xFFFFFFFFu);                    // global_addr[31:0]
    g0[3] = (unsigned)((ga >> 32) & 0x01FFFFFFu)             // global_addr[56:32]
          | 0x80000000u;                                     // type=2 ("image")
    i32x8 g1;
    g1[0] = (int)(1u << 16);                                 // data_size=1 -> 2 bytes
    g1[1] = (int)((unsigned)width << 16);                    // tensor_dim0[15:0]
    g1[2] = (int)((unsigned)height << 16);                   // dim0 hi=0 | tensor_dim1[15:0]
    g1[3] = (int)((unsigned)width << 16);                    // dim1 hi=0 | tile_dim0
    g1[4] = (int)(unsigned)height;                           // tile_dim1 | tile_dim2=0
    g1[5] = row_stride;                                      // tensor_dim0_stride[31:0]
    g1[6] = 0;                                               // stride hi | dim1_stride lo
    g1[7] = 0;
    i32x4 z4 = (i32x4)0;                                     // groups 2/3 unused (2-D)
    i32x8 z8 = (i32x8)0;
    __builtin_amdgcn_tensor_load_to_lds(g0, g1, z4, z4, z8, 0);
}
__device__ __forceinline__ unsigned lds_off_of(const void* p) {
    return (unsigned)(uintptr_t)p;                           // LDS aperture: [31:0] = offset
}
#endif

__device__ __forceinline__ unsigned short f2bf(float f) {
    union { __bf16 b; unsigned short u; } c; c.b = (__bf16)f; return c.u;
}
__device__ __forceinline__ float bf2f(unsigned short h) {
    union { unsigned short u; __bf16 b; } c; c.u = h; return (float)c.b;
}
__device__ __forceinline__ unsigned pack2bf(float lo, float hi) {
    union { __bf16 b[2]; unsigned u; } c;
    c.b[0] = (__bf16)lo; c.b[1] = (__bf16)hi;
    return c.u;
}

union AFrag { v16bf v; unsigned short h[16]; };

// A-operand (16xK row-major u16 tile, one K-chunk of 32):
// lane (l&15) = row; kb = (l>>4)*8; elems = {kb..kb+7, kb+16..kb+23}
template<int STRIDE>
__device__ __forceinline__ v16bf load_frag_rowmajor(const unsigned short* base, int lane) {
    AFrag f;
    const unsigned short* p = base + (lane & 15) * STRIDE + ((lane >> 4) << 3);
#pragma unroll
    for (int i = 0; i < 8; ++i) f.h[i] = p[i];
#pragma unroll
    for (int i = 0; i < 8; ++i) f.h[8 + i] = p[16 + i];
    return f.v;
}

// B-operand from a [K][N] row-major u16 tile (column gather).
template<int STRIDE>
__device__ __forceinline__ v16bf load_frag_colmajor(const unsigned short* base, int lane) {
    AFrag f;
    const unsigned short* p = base + (lane & 15) + (((lane >> 4) << 3) * STRIDE);
#pragma unroll
    for (int i = 0; i < 8; ++i) f.h[i] = p[i * STRIDE];
#pragma unroll
    for (int i = 0; i < 8; ++i) f.h[8 + i] = p[(16 + i) * STRIDE];
    return f.v;
}

__device__ __forceinline__ v8f wmma_bf16(v16bf a, v16bf b, v8f c) {
    return __builtin_amdgcn_wmma_f32_16x16x32_bf16(false, a, false, b, (short)0, c, false, false);
}

// stage a 128x64 fp32 tile into LDS as bf16 (float4 loads, packed converts)
__device__ __forceinline__ void stage_f32_tile(const float* __restrict__ src, int ld,
                                               unsigned short (*dst)[64], int tid) {
#pragma unroll
    for (int i = tid; i < 128 * 16; i += 256) {
        int r = i >> 4, c4 = (i & 15) << 2;
        const float4 f = *(const float4*)&src[(size_t)r * ld + c4];
        uint2 p; p.x = pack2bf(f.x, f.y); p.y = pack2bf(f.z, f.w);
        *(uint2*)&dst[r][c4] = p;
    }
}
// stage a 128x64 bf16 tile into LDS (straight b128 copy) — TDM fallback
__device__ __forceinline__ void stage_bf16_tile(const unsigned short* __restrict__ src, int ld,
                                                unsigned short (*dst)[64], int tid) {
#pragma unroll
    for (int i = tid; i < 128 * 8; i += 256) {
        int r = i >> 3, c8 = (i & 7) << 3;
        *(uint4*)&dst[r][c8] = *(const uint4*)&src[(size_t)r * ld + c8];
    }
}

// ---------------------------------------------------------------------------
// Kernel 1: qkv[m,n] = sum_k x[m,k] * w_attn[n,k] + b_attn[n]  (bf16 out)
// block tile 128x128, K-chunk 64, double-buffered LDS; 8 waves, wave 32x64.
// ---------------------------------------------------------------------------
__global__ __launch_bounds__(256)
void qkv_gemm_kernel(const float* __restrict__ X, const float* __restrict__ W,
                     const float* __restrict__ bias, unsigned short* __restrict__ out,
                     int M, int N, int K) {
    __shared__ unsigned short As[2][128][64];   // 32 KB
    __shared__ unsigned short Bs[2][128][64];   // 32 KB

    const int tid  = threadIdx.x;
    const int lane = tid & 31;
    const int wave = tid >> 5;
    const int wM   = wave >> 1;     // 0..3
    const int wN   = wave & 1;      // 0..1
    const int mblk = blockIdx.y * 128;
    const int nblk = blockIdx.x * 128;

    const float* Xb = &X[(size_t)mblk * K];
    const float* Wb = &W[(size_t)nblk * K];

    v8f acc[2][4];
#pragma unroll
    for (int i = 0; i < 2; ++i)
#pragma unroll
        for (int j = 0; j < 4; ++j) acc[i][j] = (v8f)0.0f;

    const int pfr = tid >> 1;
    const int pfc = (tid & 1) << 5;

    stage_f32_tile(Xb, K, As[0], tid);
    stage_f32_tile(Wb, K, Bs[0], tid);

    int buf = 0;
    for (int k0 = 0; k0 < K; k0 += 64, buf ^= 1) {
        __syncthreads();
        // stage NEXT chunk into the other buffer; overlaps with WMMA below
        if (k0 + 64 < K) {
            stage_f32_tile(Xb + k0 + 64, K, As[buf ^ 1], tid);
            stage_f32_tile(Wb + k0 + 64, K, Bs[buf ^ 1], tid);
        }
        if (k0 + 128 < K) {   // warm chunk after next (global_prefetch_b8)
            __builtin_prefetch(&Xb[(size_t)pfr * K + k0 + 128 + pfc], 0, 1);
            __builtin_prefetch(&Wb[(size_t)pfr * K + k0 + 128 + pfc], 0, 1);
        }

#pragma unroll
        for (int s = 0; s < 2; ++s) {
            v16bf a[2], b[4];
#pragma unroll
            for (int mt = 0; mt < 2; ++mt)
                a[mt] = load_frag_rowmajor<64>(&As[buf][wM * 32 + mt * 16][s * 32], lane);
#pragma unroll
            for (int nt = 0; nt < 4; ++nt)
                b[nt] = load_frag_rowmajor<64>(&Bs[buf][wN * 64 + nt * 16][s * 32], lane);
#pragma unroll
            for (int mt = 0; mt < 2; ++mt)
#pragma unroll
                for (int nt = 0; nt < 4; ++nt)
                    acc[mt][nt] = wmma_bf16(a[mt], b[nt], acc[mt][nt]);
        }
    }

    const int col = lane & 15;
    const int rb  = (lane >> 4) * 8;
#pragma unroll
    for (int mt = 0; mt < 2; ++mt)
#pragma unroll
        for (int nt = 0; nt < 4; ++nt) {
            int n = nblk + wN * 64 + nt * 16 + col;
            float bv = bias[n];
#pragma unroll
            for (int r = 0; r < 8; ++r) {
                int m = mblk + wM * 32 + mt * 16 + rb + r;
                out[(size_t)m * N + n] = f2bf(acc[mt][nt][r] + bv);
            }
        }
}

// ---------------------------------------------------------------------------
// Kernel 2: RoPE in-place on q,k sections of qkv [B,T,3C] (bf16)
// ---------------------------------------------------------------------------
__global__ __launch_bounds__(256)
void rope_kernel(unsigned short* __restrict__ qkv, int B_, int T, int C) {
    int idx = blockIdx.x * blockDim.x + threadIdx.x;
    int total = B_ * T * NHEAD * (HSZ / 2);
    if (idx >= total) return;
    int j = idx & 63;                 // rotation pair index 0..63
    int tmp = idx >> 6;
    int h = tmp % NHEAD; tmp /= NHEAD;
    int t = tmp % T;
    int b = tmp / T;
    float inv_freq = __expf(-__logf(10000.0f) * (float)(2 * j) * (1.0f / (float)HSZ));
    float sn, cs;
    __sincosf((float)t * inv_freq, &sn, &cs);
#pragma unroll
    for (int s = 0; s < 2; ++s) {     // 0 = q section, 1 = k section
        size_t base = (((size_t)b * T + t) * 3 + s) * C + (size_t)h * HSZ;
        float x1 = bf2f(qkv[base + j]);
        float x2 = bf2f(qkv[base + j + 64]);
        qkv[base + j]      = f2bf(x1 * cs - x2 * sn);
        qkv[base + j + 64] = f2bf(x2 * cs + x1 * sn);
    }
}

// ---------------------------------------------------------------------------
// Kernel 3: causal flash attention. One block = 128 q rows of one head.
// 8 waves; each wave owns 16 q rows, Q kept as 4 A-fragments in registers.
// KV streamed in 64-row tiles through double-buffered LDS via the TDM.
// ---------------------------------------------------------------------------
__device__ __forceinline__ void stage_kv_valu(const unsigned short* __restrict__ base,
                                              size_t rowstride, int C, int kv0,
                                              unsigned short (*Kd)[HSZ], unsigned short (*Vd)[HSZ],
                                              int tid) {
#pragma unroll
    for (int i = tid; i < 64 * 32; i += 256) {
        int r = i >> 5, d4 = (i & 31) << 2;
        size_t off = (size_t)(kv0 + r) * rowstride + d4;
        *(uint2*)&Kd[r][d4] = *(const uint2*)&base[(size_t)C + off];
        *(uint2*)&Vd[r][d4] = *(const uint2*)&base[(size_t)2 * C + off];
    }
}

__global__ __launch_bounds__(256)
void attn_kernel(const unsigned short* __restrict__ qkv, unsigned short* __restrict__ yout,
                 int B_, int T, int C) {
    __shared__ unsigned short Ks[2][64][HSZ];   // 32 KB
    __shared__ unsigned short Vs[2][64][HSZ];   // 32 KB
    __shared__ unsigned short Ps[8][16][64];    // 16 KB, per-wave P scratch

    const int tid  = threadIdx.x;
    const int lane = tid & 31;
    const int wave = tid >> 5;
    const int head = blockIdx.y;               // b * NHEAD + h
    const int b    = head / NHEAD;
    const int h    = head % NHEAD;
    const int qbase = blockIdx.x * 128;
    const int wq    = qbase + wave * 16;       // this wave's first q row

    const size_t rowstride = (size_t)3 * C;
    const unsigned short* base = qkv + (size_t)b * T * rowstride + (size_t)h * HSZ;

    // Q fragments: rows wq..wq+15, d split into 4 chunks of 32
    v16bf qf[4];
    {
        const unsigned short* qp = base + (size_t)(wq + (lane & 15)) * rowstride;
        const int kb = (lane >> 4) << 3;
#pragma unroll
        for (int c = 0; c < 4; ++c) {
            AFrag f;
            int d0 = c * 32 + kb;
#pragma unroll
            for (int i = 0; i < 8; ++i) f.h[i] = qp[d0 + i];
#pragma unroll
            for (int i = 0; i < 8; ++i) f.h[8 + i] = qp[d0 + 16 + i];
            qf[c] = f.v;
        }
    }

    v8f o[8];
#pragma unroll
    for (int j = 0; j < 8; ++j) o[j] = (v8f)0.0f;
    float mrow[8], lrow[8];
#pragma unroll
    for (int r = 0; r < 8; ++r) { mrow[r] = -1e30f; lrow[r] = 0.0f; }

    const float scale = 0.08838834764831845f;   // 1/sqrt(128)
    const int col = lane & 15;
    const int rb  = (lane >> 4) * 8;
    const int nkv = qbase / 64 + 2;             // 64-row kv tiles for causal block

    // stage first KV tile
#if HAVE_TDM
    if (tid < 32) {   // one wave issues the DMA pair
        tdm_load_2d(base + (size_t)C,     lds_off_of(&Ks[0][0][0]), HSZ, 64, (int)rowstride);
        tdm_load_2d(base + (size_t)2 * C, lds_off_of(&Vs[0][0][0]), HSZ, 64, (int)rowstride);
    }
#else
    stage_kv_valu(base, rowstride, C, 0, Ks[0], Vs[0], tid);
#endif

    int buf = 0;
    for (int it = 0; it < nkv; ++it, buf ^= 1) {
        const int kv0 = it * 64;
#if HAVE_TDM
        __builtin_amdgcn_s_wait_tensorcnt(0);   // no-op for waves with TENSORcnt==0
#endif
        __syncthreads();
        // stage next KV tile into the other buffer (overlaps with WMMA below)
        if (it + 1 < nkv) {
#if HAVE_TDM
            if (tid < 32) {
                size_t goff = (size_t)(kv0 + 64) * rowstride;
                tdm_load_2d(base + (size_t)C + goff,     lds_off_of(&Ks[buf ^ 1][0][0]),
                            HSZ, 64, (int)rowstride);
                tdm_load_2d(base + (size_t)2 * C + goff, lds_off_of(&Vs[buf ^ 1][0][0]),
                            HSZ, 64, (int)rowstride);
            }
#else
            stage_kv_valu(base, rowstride, C, kv0 + 64, Ks[buf ^ 1], Vs[buf ^ 1], tid);
#endif
        }

        if (kv0 > wq + 15) continue;            // fully masked for this wave (uniform)

        // S = Q * K^T  (four 16-wide kv subtiles)
        v8f s[4];
#pragma unroll
        for (int sub = 0; sub < 4; ++sub) {
            s[sub] = (v8f)0.0f;
#pragma unroll
            for (int c = 0; c < 4; ++c) {
                v16bf kf = load_frag_rowmajor<HSZ>(&Ks[buf][sub * 16][c * 32], lane);
                s[sub] = wmma_bf16(qf[c], kf, s[sub]);
            }
        }

        // scale + causal mask
#pragma unroll
        for (int sub = 0; sub < 4; ++sub)
#pragma unroll
            for (int r = 0; r < 8; ++r) {
                float v = s[sub][r] * scale;
                int qt = wq + rb + r;
                int kt = kv0 + sub * 16 + col;
                if (kt > qt) v = -1e30f;
                s[sub][r] = v;
            }

        // online softmax: row max / correction / exp / row sum
        float corr[8];
#pragma unroll
        for (int r = 0; r < 8; ++r) {
            float m2 = fmaxf(fmaxf(s[0][r], s[1][r]), fmaxf(s[2][r], s[3][r]));
            m2 = fmaxf(m2, __shfl_xor(m2, 1));
            m2 = fmaxf(m2, __shfl_xor(m2, 2));
            m2 = fmaxf(m2, __shfl_xor(m2, 4));
            m2 = fmaxf(m2, __shfl_xor(m2, 8));
            float mn = fmaxf(mrow[r], m2);
            corr[r] = __expf(mrow[r] - mn);
            mrow[r] = mn;
            float sum = 0.0f;
#pragma unroll
            for (int sub = 0; sub < 4; ++sub) {
                float p = __expf(s[sub][r] - mn);
                s[sub][r] = p;
                sum += p;
            }
            sum += __shfl_xor(sum, 1);
            sum += __shfl_xor(sum, 2);
            sum += __shfl_xor(sum, 4);
            sum += __shfl_xor(sum, 8);
            lrow[r] = lrow[r] * corr[r] + sum;
        }
#pragma unroll
        for (int j = 0; j < 8; ++j)
#pragma unroll
            for (int r = 0; r < 8; ++r) o[j][r] *= corr[r];

        // P (C/D layout) -> bf16 A fragments via per-wave LDS scratch
#pragma unroll
        for (int sub = 0; sub < 4; ++sub)
#pragma unroll
            for (int r = 0; r < 8; ++r)
                Ps[wave][rb + r][sub * 16 + col] = f2bf(s[sub][r]);
        v16bf pf[2];
#pragma unroll
        for (int p = 0; p < 2; ++p)
            pf[p] = load_frag_rowmajor<64>(&Ps[wave][0][p * 32], lane);

        // O += P * V  (two kv chunks of 32 per output subtile)
#pragma unroll
        for (int j = 0; j < 8; ++j) {
#pragma unroll
            for (int p = 0; p < 2; ++p) {
                v16bf vf = load_frag_colmajor<HSZ>(&Vs[buf][p * 32][j * 16], lane);
                o[j] = wmma_bf16(pf[p], vf, o[j]);
            }
        }
    }

    // epilogue: normalize and write y [B,T,C] (bf16)
    float inv_l[8];
#pragma unroll
    for (int r = 0; r < 8; ++r) inv_l[r] = 1.0f / lrow[r];
    unsigned short* yp = yout + (size_t)b * T * C + (size_t)h * HSZ;
#pragma unroll
    for (int j = 0; j < 8; ++j)
#pragma unroll
        for (int r = 0; r < 8; ++r) {
            int t = wq + rb + r;
            yp[(size_t)t * C + j * 16 + col] = f2bf(o[j][r] * inv_l[r]);
        }
}

// ---------------------------------------------------------------------------
// Kernel 4: out[m,n] = sum_k y[m,k] * w_proj[n,k] + b_proj[n]  (fp32 out)
// A tile (bf16) moved by TDM; B tile converted fp32->bf16 by VALU.
// ---------------------------------------------------------------------------
__global__ __launch_bounds__(256)
void proj_gemm_kernel(const unsigned short* __restrict__ Y, const float* __restrict__ W,
                      const float* __restrict__ bias, float* __restrict__ out,
                      int M, int N, int K) {
    __shared__ unsigned short As[2][128][64];
    __shared__ unsigned short Bs[2][128][64];

    const int tid  = threadIdx.x;
    const int lane = tid & 31;
    const int wave = tid >> 5;
    const int wM   = wave >> 1;
    const int wN   = wave & 1;
    const int mblk = blockIdx.y * 128;
    const int nblk = blockIdx.x * 128;

    const unsigned short* Yb = &Y[(size_t)mblk * K];
    const float* Wb = &W[(size_t)nblk * K];

    v8f acc[2][4];
#pragma unroll
    for (int i = 0; i < 2; ++i)
#pragma unroll
        for (int j = 0; j < 4; ++j) acc[i][j] = (v8f)0.0f;

    const int pfr = tid >> 1;
    const int pfc = (tid & 1) << 5;

#if HAVE_TDM
    if (tid < 32)
        tdm_load_2d(Yb, lds_off_of(&As[0][0][0]), 64, 128, K);
#else
    stage_bf16_tile(Yb, K, As[0], tid);
#endif
    stage_f32_tile(Wb, K, Bs[0], tid);

    int buf = 0;
    for (int k0 = 0; k0 < K; k0 += 64, buf ^= 1) {
#if HAVE_TDM
        __builtin_amdgcn_s_wait_tensorcnt(0);
#endif
        __syncthreads();
        if (k0 + 64 < K) {
#if HAVE_TDM
            if (tid < 32)
                tdm_load_2d(Yb + k0 + 64, lds_off_of(&As[buf ^ 1][0][0]), 64, 128, K);
#else
            stage_bf16_tile(Yb + k0 + 64, K, As[buf ^ 1], tid);
#endif
            stage_f32_tile(Wb + k0 + 64, K, Bs[buf ^ 1], tid);
        }
        if (k0 + 128 < K) {
            __builtin_prefetch(&Yb[(size_t)pfr * K + k0 + 128], 0, 1);
            __builtin_prefetch(&Wb[(size_t)pfr * K + k0 + 128 + pfc], 0, 1);
        }

#pragma unroll
        for (int s = 0; s < 2; ++s) {
            v16bf a[2], b[4];
#pragma unroll
            for (int mt = 0; mt < 2; ++mt)
                a[mt] = load_frag_rowmajor<64>(&As[buf][wM * 32 + mt * 16][s * 32], lane);
#pragma unroll
            for (int nt = 0; nt < 4; ++nt)
                b[nt] = load_frag_rowmajor<64>(&Bs[buf][wN * 64 + nt * 16][s * 32], lane);
#pragma unroll
            for (int mt = 0; mt < 2; ++mt)
#pragma unroll
                for (int nt = 0; nt < 4; ++nt)
                    acc[mt][nt] = wmma_bf16(a[mt], b[nt], acc[mt][nt]);
        }
    }

    const int col = lane & 15;
    const int rb  = (lane >> 4) * 8;
#pragma unroll
    for (int mt = 0; mt < 2; ++mt)
#pragma unroll
        for (int nt = 0; nt < 4; ++nt) {
            int n = nblk + wN * 64 + nt * 16 + col;
            float bv = bias[n];
#pragma unroll
            for (int r = 0; r < 8; ++r) {
                int m = mblk + wM * 32 + mt * 16 + rb + r;
                out[(size_t)m * N + n] = acc[mt][nt][r] + bv;
            }
        }
}

// ---------------------------------------------------------------------------
extern "C" void kernel_launch(void* const* d_in, const int* in_sizes, int n_in,
                              void* d_out, int out_size, void* d_ws, size_t ws_size,
                              hipStream_t stream) {
    const float* x      = (const float*)d_in[0];
    const float* w_attn = (const float*)d_in[1];
    const float* b_attn = (const float*)d_in[2];
    const float* w_proj = (const float*)d_in[3];
    const float* b_proj = (const float*)d_in[4];
    float* out = (float*)d_out;

    const int B_ = 2, T = 2048, C = 2048;
    const int M = B_ * T;           // 4096

    // workspace: qkv bf16 [B,T,3C] (48 MiB) + y bf16 [B,T,C] (16 MiB)
    unsigned short* qkv = (unsigned short*)d_ws;
    unsigned short* yb  = qkv + (size_t)M * 3 * C;

    // 1) fused QKV projection (bf16 WMMA, fp32 accumulate)
    {
        dim3 grid(3 * C / 128, M / 128);
        qkv_gemm_kernel<<<grid, 256, 0, stream>>>(x, w_attn, b_attn, qkv, M, 3 * C, C);
    }
    // 2) RoPE in place on q,k
    {
        int total = B_ * T * NHEAD * (HSZ / 2);
        rope_kernel<<<(total + 255) / 256, 256, 0, stream>>>(qkv, B_, T, C);
    }
    // 3) causal flash attention
    {
        dim3 grid(T / 128, B_ * NHEAD);
        attn_kernel<<<grid, 256, 0, stream>>>(qkv, yb, B_, T, C);
    }
    // 4) output projection (fp32 out + bias)
    {
        dim3 grid(C / 128, M / 128);
        proj_gemm_kernel<<<grid, 256, 0, stream>>>(yb, w_proj, b_proj, out, M, C, C);
    }
}
